// ABLTagger_39986145525802
// MI455X (gfx1250) — matrix operand order repaired and checked by reading, not compiled, not run on hardware
//
#include <hip/hip_runtime.h>
#include <hip/hip_bf16.h>

// ---------------------------------------------------------------------------
// ABLTagger on gfx1250: all matmuls via v_wmma_f32_16x16x32_bf16 (wave32),
// recurrent weights staged into LDS via the Tensor Data Mover.
// ---------------------------------------------------------------------------

typedef __bf16 bf16_t;
typedef __attribute__((ext_vector_type(16))) __bf16 v16bf;
typedef __attribute__((ext_vector_type(8)))  float  v8f;
typedef __attribute__((ext_vector_type(4)))  unsigned int v4u;
typedef __attribute__((ext_vector_type(8)))  int v8i;
typedef __attribute__((ext_vector_type(4)))  int v4i;

#define DEV __device__ __forceinline__

#define BB 128
#define SS 128
#define CC 16
#define EMBC 20
#define NTOK (BB * SS)        // 16384
#define TAGS 565

// ---- workspace layout (bytes) ----
#define OFF_CWIF  ((size_t)0)          // 16*1*512*2  = 16384
#define OFF_CWIB  ((size_t)16384)      // 16384
#define OFF_CWHF  ((size_t)32768)      // 16*2*512*2  = 32768
#define OFF_MWIF  ((size_t)65536)      // 16*11*512*2 = 180224
#define OFF_MWIB  ((size_t)245760)     // 180224
#define OFF_MWHF  ((size_t)425984)     // 32768
#define OFF_MWHB  ((size_t)458752)     // 32768
#define OFF_W1    ((size_t)491520)     // 2*4*512*2   = 8192
#define OFF_W2    ((size_t)499712)     // 36*1*512*2  = 36864
#define OFF_CAW   ((size_t)(1u << 20))                   // f32 NTOK*128  (8 MB)
#define OFF_XG    (OFF_CAW + (size_t)NTOK * 128 * 4)     // f32 2*NTOK*256 (32 MB)
#define OFF_MOUT  (OFF_XG + (size_t)2 * NTOK * 256 * 4)  // f32 NTOK*128 (8 MB)

DEV float fast_sig(float x)  { return 1.0f / (1.0f + __expf(-x)); }
DEV float fast_tanh(float x) { return 1.0f - 2.0f / (__expf(2.0f * x) + 1.0f); }

DEV v8f v8zero() { v8f z;
#pragma unroll
  for (int i = 0; i < 8; ++i) z[i] = 0.0f;
  return z; }

DEV v8f v8splat(float x) { v8f z;
#pragma unroll
  for (int i = 0; i < 8; ++i) z[i] = x;
  return z; }

// Opaque zero: blocks LICM from hoisting loop-invariant fragment loads
// (prevents the hoist->spill->scratch-reload pattern seen in round 1).
DEV int opq() { int x = 0; asm volatile("" : "+s"(x)); return x; }

// A fragment (16x32 bf16, row-major source in LDS, leading dim ldk).
// Lane L<16 : row M=L,    K = {0..7, 16..23}
// Lane L>=16: row M=L-16, K = {8..15, 24..31}
DEV v16bf load_A(const bf16_t* base, int ldk, int k0) {
  int lane = threadIdx.x & 31;
  int row  = lane & 15;
  int kh   = (lane >> 4) & 1;
  const bf16_t* p = base + row * ldk + k0 + kh * 8;
  v16bf a;
#pragma unroll
  for (int v = 0; v < 4; ++v) {
    a[2 * v]     = p[2 * v];
    a[2 * v + 1] = p[2 * v + 1];
    a[8 + 2 * v] = p[16 + 2 * v];
    a[9 + 2 * v] = p[16 + 2 * v + 1];
  }
  return a;
}

// B fragment (lane-major packed fragments of 512 bf16), global or LDS source.
DEV v16bf load_B(const bf16_t* packed, int ntile, int ktiles, int ktile) {
  int lane = threadIdx.x & 31;
  const v16bf* f = (const v16bf*)packed;
  return f[((ntile * ktiles + ktile) << 5) + lane];
}

DEV v8f wmma_bf16(v16bf a, v16bf b, v8f c) {
  return __builtin_amdgcn_wmma_f32_16x16x32_bf16(false, a, false, b, (short)0, c,
                                                 false, false);
}

// ---------------------------------------------------------------------------
// Tensor Data Mover: 1-D DMA of `bytes` (multiple of 8) global -> LDS.
// D# group0: count=1, lds_addr, global_addr[56:0], type=2 ("image").
// D# group1: data_size=3 (8B), tensor_dim0 = tile_dim0 = bytes/8, 1-D tile.
// This toolchain exposes the 6-arg builtin (extra int32x8 group).
// ---------------------------------------------------------------------------
DEV void tdm_load_to_lds(void* lds_dst, const void* gsrc, unsigned bytes) {
  unsigned lds_off = (unsigned)(size_t)lds_dst;   // addr[31:0] = LDS offset
  unsigned long long ga = (unsigned long long)(size_t)gsrc;
  unsigned n8 = bytes >> 3;                       // 8-byte elements
  v4u g0;
  g0[0] = 1u;                                     // count=1, user descriptor
  g0[1] = lds_off;                                // lds_addr  (bits 63:32)
  g0[2] = (unsigned)(ga & 0xffffffffu);           // global_addr lo (95:64)
  g0[3] = (unsigned)((ga >> 32) & 0x01ffffffu) | (2u << 30);  // hi + type=2
  v8i g1;
  g1[0] = (int)(3u << 16);                        // wg_mask=0, data_size=8B
  g1[1] = (int)((n8 & 0xffffu) << 16);            // tensor_dim0[15:0] @63:48
  g1[2] = (int)((n8 >> 16) & 0xffffu);            // tensor_dim0[31:16] @79:64
  g1[3] = (int)((n8 & 0xffffu) << 16);            // tile_dim0 @127:112
  g1[4] = 0;                                      // tile_dim1/2 unused
  g1[5] = (int)n8;                                // tensor_dim0_stride lo
  g1[6] = 0;
  g1[7] = 0;
  v4i z4 = {0, 0, 0, 0};
  v8i z8 = {0, 0, 0, 0, 0, 0, 0, 0};
  __builtin_amdgcn_tensor_load_to_lds(g0, g1, z4, z4, z8, 0);
}

// ---------------------------------------------------------------------------
// Pack W (Norig x Korig, row-major f32) into WMMA B fragments:
// frag(ntile,ktile)[lane][i]: n = ntile*16+(lane&15), k = ktile*32+(i>>3)*16+(lane>>4)*8+(i&7)
// ---------------------------------------------------------------------------
__global__ void k_pack(const float* __restrict__ W, bf16_t* __restrict__ dst,
                       int Norig, int Korig, int ntiles, int ktiles) {
  int total = ntiles * ktiles * 512;
  for (int e = blockIdx.x * blockDim.x + threadIdx.x; e < total;
       e += gridDim.x * blockDim.x) {
    int frag = e >> 9;
    int lane = (e >> 4) & 31;
    int i    = e & 15;
    int ktile = frag % ktiles;
    int ntile = frag / ktiles;
    int n = ntile * 16 + (lane & 15);
    int k = ktile * 32 + ((i >> 3) << 4) + ((lane >> 4) << 3) + (i & 7);
    float v = (n < Norig && k < Korig) ? W[(size_t)n * Korig + k] : 0.0f;
    dst[e] = (bf16_t)v;
  }
}

// ---------------------------------------------------------------------------
// Char-level BiLSTM -> chars_as_word (NTOK x 128).
// 4 waves/block, one 16-token group per wave; input projection fused into the
// 16-step loop; weights staged in LDS by the TDM. Backward direction = single
// step with h=c=0 (the reference's cout[clen-1] maps to bwd step 0).
// ---------------------------------------------------------------------------
__global__ __launch_bounds__(128) void k_char(
    const int* __restrict__ inp, const float* __restrict__ Echar,
    const float* __restrict__ cbf, const float* __restrict__ cbb,
    const bf16_t* __restrict__ wif, const bf16_t* __restrict__ whf,
    const bf16_t* __restrict__ wib, float* __restrict__ caw) {
  __shared__ __align__(32) bf16_t s_wif[16 * 1 * 512];   // 16 KB
  __shared__ __align__(32) bf16_t s_whf[16 * 2 * 512];   // 32 KB
  __shared__ __align__(32) bf16_t s_ce[4][16 * 32];
  __shared__ __align__(32) bf16_t s_h[4][16 * 64];
  __shared__ int s_cl[4][16];
  int wave = threadIdx.x >> 5;
  int lane = threadIdx.x & 31;
  int grp  = blockIdx.x * 4 + wave;
  int row  = lane & 15, kh = lane >> 4, col = row;
  int tok  = grp * 16 + row;
  bf16_t* ce = s_ce[wave];
  bf16_t* hh = s_h[wave];

  if (wave == 0) {                      // one wave DMAs the shared weights
    tdm_load_to_lds(s_wif, wif, 16 * 512 * 2);
    tdm_load_to_lds(s_whf, whf, 32 * 512 * 2);
    __builtin_amdgcn_s_wait_tensorcnt(0);
  }

  int ids[16];
  int clen = 0;
#pragma unroll
  for (int t = 0; t < CC; ++t) {
    ids[t] = inp[tok * 19 + t];
    clen += (ids[t] != 0);
  }
  if (kh == 0) s_cl[wave][row] = clen;
#pragma unroll
  for (int k = 0; k < 6; ++k) ce[row * 32 + 20 + kh * 6 + k] = (bf16_t)0.0f;
#pragma unroll
  for (int k = 0; k < 32; ++k) hh[lane * 32 + k] = (bf16_t)0.0f;

  __syncthreads();                      // weights staged + s_cl visible

  int cl8[8];
#pragma unroll
  for (int r = 0; r < 8; ++r) cl8[r] = s_cl[wave][kh * 8 + r];

  v8f hr[4], cr[4];
#pragma unroll
  for (int q = 0; q < 4; ++q) { hr[q] = v8zero(); cr[q] = v8zero(); }

  // ---- forward direction: 16 masked steps, keep final h ----
  for (int t = 0; t < CC; ++t) {
    {
      const float* e = Echar + (size_t)ids[t] * EMBC;
#pragma unroll
      for (int k = 0; k < 10; ++k)
        ce[row * 32 + kh * 10 + k] = (bf16_t)e[kh * 10 + k];
    }
    const bf16_t* wifp = s_wif + opq();   // defeat LICM hoist/spill
    const bf16_t* whfp = s_whf + opq();
    v16bf a_in = load_A(ce, 32, 0);
    v16bf a_h0 = load_A(hh, 64, 0);
    v16bf a_h1 = load_A(hh, 64, 32);
#pragma unroll
    for (int cg = 0; cg < 4; ++cg) {
      v8f g4[4];
#pragma unroll
      for (int gi = 0; gi < 4; ++gi) {
        int nt = gi * 4 + cg;                     // i,f,g,o blocks of 4 tiles
        v8f acc = v8splat(cbf[nt * 16 + col]);
        acc = wmma_bf16(a_in, load_B(wifp, nt, 1, 0), acc);
        acc = wmma_bf16(a_h0, load_B(whfp, nt, 2, 0), acc);
        acc = wmma_bf16(a_h1, load_B(whfp, nt, 2, 1), acc);
        g4[gi] = acc;
      }
#pragma unroll
      for (int r = 0; r < 8; ++r) {
        float i_ = fast_sig(g4[0][r]);
        float f_ = fast_sig(g4[1][r]);
        float gg = fast_tanh(g4[2][r]);
        float o_ = fast_sig(g4[3][r]);
        float cn = f_ * cr[cg][r] + i_ * gg;
        float hn = o_ * fast_tanh(cn);
        bool  mt = t < cl8[r];
        cr[cg][r] = mt ? cn : cr[cg][r];
        hr[cg][r] = mt ? hn : hr[cg][r];
      }
#pragma unroll
      for (int r = 0; r < 8; ++r)
        hh[(kh * 8 + r) * 64 + cg * 16 + col] = (bf16_t)hr[cg][r];
    }
  }
#pragma unroll
  for (int cg = 0; cg < 4; ++cg)
#pragma unroll
    for (int r = 0; r < 8; ++r)
      caw[(size_t)(grp * 16 + kh * 8 + r) * 128 + cg * 16 + col] = hr[cg][r];

  // ---- backward direction: single step on last valid char, h=c=0 ----
  {
    int pos = clen > 0 ? clen - 1 : 0;
    const float* e = Echar + (size_t)ids[pos] * EMBC;
#pragma unroll
    for (int k = 0; k < 10; ++k)
      ce[row * 32 + kh * 10 + k] = (bf16_t)e[kh * 10 + k];
    const bf16_t* wibp = wib + opq();
    v16bf a_in = load_A(ce, 32, 0);
#pragma unroll
    for (int cg = 0; cg < 4; ++cg) {
      v8f g4[4];
#pragma unroll
      for (int gi = 0; gi < 4; ++gi) {
        int nt = gi * 4 + cg;
        v8f acc = v8splat(cbb[nt * 16 + col]);
        g4[gi] = wmma_bf16(a_in, load_B(wibp, nt, 1, 0), acc);
      }
#pragma unroll
      for (int r = 0; r < 8; ++r) {
        float i_ = fast_sig(g4[0][r]);
        float gg = fast_tanh(g4[2][r]);
        float o_ = fast_sig(g4[3][r]);
        float hn = o_ * fast_tanh(i_ * gg);
        float y  = (cl8[r] > 0) ? hn : 0.0f;
        caw[(size_t)(grp * 16 + kh * 8 + r) * 128 + 64 + cg * 16 + col] = y;
      }
    }
  }
}

// ---------------------------------------------------------------------------
// Main-LSTM input projection: xg[dir][tok][256] = main_in @ Wi.T + b
// main_in = [caw(128) | E_tok(128) | E_morph(64) | E_ctag(32)] = 352 = 11*32.
// ---------------------------------------------------------------------------
__global__ __launch_bounds__(128) void k_main_gates(
    const int* __restrict__ inp, const float* __restrict__ caw,
    const float* __restrict__ Etok, const float* __restrict__ Emorph,
    const float* __restrict__ Ectag, const float* __restrict__ mbf,
    const float* __restrict__ mbb, const bf16_t* __restrict__ wif,
    const bf16_t* __restrict__ wib, float* __restrict__ xg) {
  __shared__ __align__(32) bf16_t s_in[4][16 * 352];
  int wave = threadIdx.x >> 5, lane = threadIdx.x & 31;
  int grp  = blockIdx.x * 4 + wave;
  int row  = lane & 15, kh = lane >> 4, col = row;
  int tok  = grp * 16 + row;
  bf16_t* xin = s_in[wave];
  int w  = inp[tok * 19 + 16];
  int m  = inp[tok * 19 + 17];
  int ct = inp[tok * 19 + 18];
  {
    const float* cawr = caw + (size_t)tok * 128;
    const float* et   = Etok + (size_t)w * 128;
    const float* em   = Emorph + (size_t)m * 64;
    const float* ec   = Ectag + (size_t)ct * 32;
    for (int j = 0; j < 176; ++j) {
      int k = kh * 176 + j;
      float v;
      if (k < 128)      v = cawr[k];
      else if (k < 256) v = et[k - 128];
      else if (k < 320) v = em[k - 256];
      else              v = ec[k - 320];
      xin[row * 352 + k] = (bf16_t)v;
    }
  }
  v16bf A[11];
#pragma unroll
  for (int kt = 0; kt < 11; ++kt) A[kt] = load_A(xin, 352, kt * 32);
  for (int dir = 0; dir < 2; ++dir) {
    const bf16_t* wi  = dir ? wib : wif;
    const float* bias = dir ? mbb : mbf;
    float* xgd = xg + (size_t)dir * NTOK * 256;
    for (int nt = 0; nt < 16; ++nt) {
      const bf16_t* wip = wi + opq();   // keep fragment loads inside the loop
      v8f acc = v8splat(bias[nt * 16 + col]);
#pragma unroll
      for (int kt = 0; kt < 11; ++kt)
        acc = wmma_bf16(A[kt], load_B(wip, nt, 11, kt), acc);
#pragma unroll
      for (int r = 0; r < 8; ++r)
        xgd[(size_t)(grp * 16 + kh * 8 + r) * 256 + nt * 16 + col] = acc[r];
    }
  }
}

// ---------------------------------------------------------------------------
// Main BiLSTM recurrence. 16 blocks = {dir} x {8 batch tiles of 16 sentences}.
// Recurrent weights (32 KB) staged in LDS by the TDM; backward direction reads
// xg / writes output with _reverse_padded indexing.
// ---------------------------------------------------------------------------
__global__ __launch_bounds__(32) void k_main_lstm(
    const int* __restrict__ inp, const float* __restrict__ xg,
    const bf16_t* __restrict__ whf, const bf16_t* __restrict__ whb,
    float* __restrict__ mout) {
  __shared__ __align__(32) bf16_t s_wb[16 * 2 * 512];   // 32 KB
  __shared__ __align__(32) bf16_t s_h[16 * 64];
  int lane = threadIdx.x & 31;
  int dir  = blockIdx.x >> 3;
  int mt8  = blockIdx.x & 7;
  int kh = lane >> 4, col = lane & 15;
  const bf16_t* wh = dir ? whb : whf;
  const float* xgd = xg + (size_t)dir * NTOK * 256;

  tdm_load_to_lds(s_wb, wh, 32 * 512 * 2);   // single wave: DMA then wait

  int sb[8], sl[8];
#pragma unroll
  for (int r = 0; r < 8; ++r) {
    sb[r] = mt8 * 16 + kh * 8 + r;
    int c = 0;
    for (int s = 0; s < SS; ++s)
      c += (inp[(size_t)(sb[r] * SS + s) * 19 + 16] != 0);
    sl[r] = c;
  }
  for (int k = 0; k < 32; ++k) s_h[lane * 32 + k] = (bf16_t)0.0f;
  v8f hr[4], cr[4];
#pragma unroll
  for (int q = 0; q < 4; ++q) { hr[q] = v8zero(); cr[q] = v8zero(); }

  __builtin_amdgcn_s_wait_tensorcnt(0);

  for (int t = 0; t < SS; ++t) {
    int tt[8];
#pragma unroll
    for (int r = 0; r < 8; ++r)
      tt[r] = (dir == 0) ? t : ((t < sl[r]) ? sl[r] - 1 - t : t);
    if (t + 1 < SS) {                     // prefetch next step's gate row
      int tn = (dir == 0) ? t + 1 : ((t + 1 < sl[0]) ? sl[0] - 2 - t : t + 1);
      __builtin_prefetch(xgd + (size_t)(sb[0] * SS + tn) * 256, 0, 1);
    }
    const bf16_t* whp = s_wb + opq();     // defeat LICM hoist/spill
    v16bf a0 = load_A(s_h, 64, 0);
    v16bf a1 = load_A(s_h, 64, 32);
#pragma unroll
    for (int cg = 0; cg < 4; ++cg) {
      v8f g4[4];
#pragma unroll
      for (int gi = 0; gi < 4; ++gi) {
        int nt = gi * 4 + cg;
        v8f acc;
#pragma unroll
        for (int r = 0; r < 8; ++r)
          acc[r] = xgd[(size_t)(sb[r] * SS + tt[r]) * 256 + nt * 16 + col];
        acc = wmma_bf16(a0, load_B(whp, nt, 2, 0), acc);
        acc = wmma_bf16(a1, load_B(whp, nt, 2, 1), acc);
        g4[gi] = acc;
      }
#pragma unroll
      for (int r = 0; r < 8; ++r) {
        float i_ = fast_sig(g4[0][r]);
        float f_ = fast_sig(g4[1][r]);
        float gg = fast_tanh(g4[2][r]);
        float o_ = fast_sig(g4[3][r]);
        float cn = f_ * cr[cg][r] + i_ * gg;
        float hn = o_ * fast_tanh(cn);
        bool  mt = t < sl[r];
        cr[cg][r] = mt ? cn : cr[cg][r];
        hr[cg][r] = mt ? hn : hr[cg][r];
        float y  = mt ? hr[cg][r] : 0.0f;
        int  pos = (dir == 0) ? t : tt[r];
        mout[(size_t)(sb[r] * SS + pos) * 128 + dir * 64 + cg * 16 + col] = y;
      }
#pragma unroll
      for (int r = 0; r < 8; ++r)
        s_h[(kh * 8 + r) * 64 + cg * 16 + col] = (bf16_t)hr[cg][r];
    }
  }
}

// ---------------------------------------------------------------------------
// Head: out = tanh(mout @ W1.T) @ W2.T   (128 -> 32 -> 565, N padded to 576)
// ---------------------------------------------------------------------------
__global__ __launch_bounds__(256) void k_head(
    const float* __restrict__ mout, const bf16_t* __restrict__ w1,
    const bf16_t* __restrict__ w2, float* __restrict__ out) {
  __shared__ __align__(32) bf16_t s_a[8][16 * 128];
  __shared__ __align__(32) bf16_t s_hid[8][16 * 32];
  int wave = threadIdx.x >> 5, lane = threadIdx.x & 31;
  int grp  = blockIdx.x * 8 + wave;
  int row  = lane & 15, kh = lane >> 4, col = row;
  bf16_t* sa = s_a[wave];
  bf16_t* sh = s_hid[wave];
  {
    const float* src = mout + (size_t)(grp * 16 + row) * 128;
    for (int j = 0; j < 64; ++j)
      sa[row * 128 + kh * 64 + j] = (bf16_t)src[kh * 64 + j];
  }
  v16bf A1[4];
#pragma unroll
  for (int kt = 0; kt < 4; ++kt) A1[kt] = load_A(sa, 128, kt * 32);
#pragma unroll
  for (int nt = 0; nt < 2; ++nt) {
    v8f acc = v8zero();
#pragma unroll
    for (int kt = 0; kt < 4; ++kt)
      acc = wmma_bf16(A1[kt], load_B(w1, nt, 4, kt), acc);
#pragma unroll
    for (int r = 0; r < 8; ++r)
      sh[(kh * 8 + r) * 32 + nt * 16 + col] = (bf16_t)fast_tanh(acc[r]);
  }
  v16bf A2 = load_A(sh, 32, 0);
  for (int nt = 0; nt < 36; ++nt) {
    const bf16_t* w2p = w2 + opq();       // keep fragment loads inside loop
    v8f acc = wmma_bf16(A2, load_B(w2p, nt, 1, 0), v8zero());
    int n = nt * 16 + col;
    if (n < TAGS) {
#pragma unroll
      for (int r = 0; r < 8; ++r)
        out[(size_t)(grp * 16 + kh * 8 + r) * TAGS + n] = acc[r];
    }
  }
}

// ---------------------------------------------------------------------------
extern "C" void kernel_launch(void* const* d_in, const int* in_sizes, int n_in,
                              void* d_out, int out_size, void* d_ws,
                              size_t ws_size, hipStream_t stream) {
  const int*   inp    = (const int*)d_in[0];
  const float* Echar  = (const float*)d_in[1];
  const float* Etok   = (const float*)d_in[2];
  const float* Emorph = (const float*)d_in[3];
  const float* Ectag  = (const float*)d_in[4];
  const float* cWif   = (const float*)d_in[5];
  const float* cWhf   = (const float*)d_in[6];
  const float* cbf    = (const float*)d_in[7];
  const float* cWib   = (const float*)d_in[8];
  /* d_in[9] = cWhb: unused — backward char LSTM contributes only step 0 (h=0) */
  const float* cbb    = (const float*)d_in[10];
  const float* mWif   = (const float*)d_in[11];
  const float* mWhf   = (const float*)d_in[12];
  const float* mbf    = (const float*)d_in[13];
  const float* mWib   = (const float*)d_in[14];
  const float* mWhb   = (const float*)d_in[15];
  const float* mbb    = (const float*)d_in[16];
  const float* W1     = (const float*)d_in[17];
  const float* W2     = (const float*)d_in[18];

  char* ws = (char*)d_ws;
  bf16_t* p_cwif = (bf16_t*)(ws + OFF_CWIF);
  bf16_t* p_cwib = (bf16_t*)(ws + OFF_CWIB);
  bf16_t* p_cwhf = (bf16_t*)(ws + OFF_CWHF);
  bf16_t* p_mwif = (bf16_t*)(ws + OFF_MWIF);
  bf16_t* p_mwib = (bf16_t*)(ws + OFF_MWIB);
  bf16_t* p_mwhf = (bf16_t*)(ws + OFF_MWHF);
  bf16_t* p_mwhb = (bf16_t*)(ws + OFF_MWHB);
  bf16_t* p_w1   = (bf16_t*)(ws + OFF_W1);
  bf16_t* p_w2   = (bf16_t*)(ws + OFF_W2);
  float*  caw    = (float*)(ws + OFF_CAW);
  float*  xg     = (float*)(ws + OFF_XG);
  float*  mout   = (float*)(ws + OFF_MOUT);

  // weight packing into WMMA B fragments (bf16, K padded to multiples of 32)
  k_pack<<<64, 256, 0, stream>>>(cWif, p_cwif, 256, 20, 16, 1);
  k_pack<<<64, 256, 0, stream>>>(cWib, p_cwib, 256, 20, 16, 1);
  k_pack<<<64, 256, 0, stream>>>(cWhf, p_cwhf, 256, 64, 16, 2);
  k_pack<<<64, 256, 0, stream>>>(mWif, p_mwif, 256, 352, 16, 11);
  k_pack<<<64, 256, 0, stream>>>(mWib, p_mwib, 256, 352, 16, 11);
  k_pack<<<64, 256, 0, stream>>>(mWhf, p_mwhf, 256, 64, 16, 2);
  k_pack<<<64, 256, 0, stream>>>(mWhb, p_mwhb, 256, 64, 16, 2);
  k_pack<<<64, 256, 0, stream>>>(W1, p_w1, 32, 128, 2, 4);
  k_pack<<<64, 256, 0, stream>>>(W2, p_w2, TAGS, 32, 36, 1);

  // char BiLSTM -> chars_as_word
  k_char<<<256, 128, 0, stream>>>(inp, Echar, cbf, cbb, p_cwif, p_cwhf, p_cwib,
                                  caw);
  // main input projection (both directions)
  k_main_gates<<<256, 128, 0, stream>>>(inp, caw, Etok, Emorph, Ectag, mbf, mbb,
                                        p_mwif, p_mwib, xg);
  // main BiLSTM recurrence
  k_main_lstm<<<16, 32, 0, stream>>>(inp, xg, p_mwhf, p_mwhb, mout);
  // MLP head
  k_head<<<128, 256, 0, stream>>>(mout, p_w1, p_w2, (float*)d_out);
}
